// MultiHeadAttentionNaive_10342281248971
// MI455X (gfx1250) — compile-verified, matrix operand order checked by reading, not run on hardware
//
#include <hip/hip_runtime.h>

// ---------------- types for WMMA fragments ----------------
typedef __attribute__((ext_vector_type(8)))  __bf16 v8bf;
typedef __attribute__((ext_vector_type(16))) __bf16 v16bf;
typedef __attribute__((ext_vector_type(8)))  float  v8f;

// ---------------- problem constants ----------------
constexpr int BATCH   = 4;
constexpr int SEQ     = 2048;
constexpr int DMODEL  = 1024;
constexpr int NHEAD   = 16;
constexpr int DHEAD   = 64;
constexpr int MROWS   = BATCH * SEQ;        // 8192
constexpr int NQKV    = 3 * DMODEL;         // 3072

// ---------------- workspace layout (bytes) ----------------
constexpr size_t OFF_X     = 0;                                        // [8192,1024] bf16
constexpr size_t OFF_WQKVT = OFF_X     + (size_t)MROWS * DMODEL * 2;   // [3072,1024] bf16 (Wqkv^T)
constexpr size_t OFF_WOUTT = OFF_WQKVT + (size_t)NQKV * DMODEL * 2;    // [1024,1024] bf16 (Wout^T)
constexpr size_t OFF_Q     = OFF_WOUTT + (size_t)DMODEL * DMODEL * 2;  // [64,2048,64] bf16 (pre-scaled)
constexpr size_t OFF_K     = OFF_Q     + (size_t)64 * SEQ * DHEAD * 2; // [64,2048,64] bf16
constexpr size_t OFF_V     = OFF_K     + (size_t)64 * SEQ * DHEAD * 2; // [64,64,2048] bf16 (V^T per head)
constexpr size_t OFF_A     = OFF_V     + (size_t)64 * DHEAD * SEQ * 2; // [8192,1024] bf16 attn out

// ---------------- fragment helpers ----------------
// 16-bit A/B fragment: lane holds two contiguous 8-element (16B) chunks.
// lanes 0-15 : K-offsets {0..7, 16..23};  lanes 16-31 : {8..15, 24..31}.
__device__ __forceinline__ v16bf ldfrag(const __bf16* p, int off0) {
  v8bf lo = *reinterpret_cast<const v8bf*>(p + off0);
  v8bf hi = *reinterpret_cast<const v8bf*>(p + off0 + 16);
  v16bf r;
#pragma unroll
  for (int i = 0; i < 8; ++i) { r[i] = lo[i]; r[i + 8] = hi[i]; }
  return r;
}

__device__ __forceinline__ v8f wmma_bf16(v16bf a, v16bf b, v8f c) {
  return __builtin_amdgcn_wmma_f32_16x16x32_bf16(
      /*neg_a=*/false, a, /*neg_b=*/false, b,
      /*c_mod=*/(short)0, c, /*reuse_a=*/false, /*reuse_b=*/false);
}

__device__ __forceinline__ v8f vzero8() {
  v8f z;
#pragma unroll
  for (int i = 0; i < 8; ++i) z[i] = 0.0f;
  return z;
}

// ---------------- prep kernels ----------------
__global__ void cvt_f32_bf16(const float* __restrict__ src, __bf16* __restrict__ dst, int n) {
  int i = blockIdx.x * blockDim.x + threadIdx.x;
  if (i < n) dst[i] = (__bf16)src[i];
}

// dst[n*K + k] = src[k*Ncols + n]   (weight transpose + convert; K = 1024)
__global__ void transpose_bf16(const float* __restrict__ src, __bf16* __restrict__ dst,
                               int K, int Ncols, int total) {
  int i = blockIdx.x * blockDim.x + threadIdx.x;
  if (i >= total) return;
  int n = i / K;
  int k = i - n * K;
  dst[i] = (__bf16)src[(size_t)k * Ncols + n];
}

// ---------------- shared 32x64-tile GEMM mainloop ----------------
// acc[g][t]: row-group g (16 rows each), col-tile t (16 cols each).
// 12 x 16B loads per K-step of 32 feed 8 WMMAs.
__device__ __forceinline__ void gemm32x64(const __bf16* __restrict__ A0,
                                          const __bf16* __restrict__ A1,
                                          const __bf16* const br[4],
                                          int off0, v8f acc[2][4]) {
#pragma unroll 2
  for (int k0 = 0; k0 < DMODEL; k0 += 32) {
    __builtin_prefetch(A0 + k0 + 256, 0, 1);
    __builtin_prefetch(A1 + k0 + 256, 0, 1);
    v16bf a0 = ldfrag(A0 + k0, off0);
    v16bf a1 = ldfrag(A1 + k0, off0);
#pragma unroll
    for (int t = 0; t < 4; ++t) {
      v16bf b = ldfrag(br[t] + k0, off0);
      acc[0][t] = wmma_bf16(a0, b, acc[0][t]);
      acc[1][t] = wmma_bf16(a1, b, acc[1][t]);
    }
  }
}

// ---------------- QKV projection: [8192,1024] @ Wqkv + b, scatter per-head ----------------
__global__ __launch_bounds__(256) void qkv_gemm(
    const __bf16* __restrict__ X, const __bf16* __restrict__ Wt,
    const float* __restrict__ bias,
    __bf16* __restrict__ Qb, __bf16* __restrict__ Kb, __bf16* __restrict__ Vt) {
  const int lane = threadIdx.x & 31;
  const int gw   = blockIdx.x * 8 + (threadIdx.x >> 5);
  const int TN   = NQKV / 64;                 // 48
  const int tm   = gw / TN, tn = gw - tm * TN;
  const int l15  = lane & 15;
  const int off0 = (lane < 16) ? 0 : 8;

  v8f acc[2][4];
#pragma unroll
  for (int g = 0; g < 2; ++g)
#pragma unroll
    for (int t = 0; t < 4; ++t) acc[g][t] = vzero8();

  const __bf16* a0 = X + (size_t)(tm * 32 + l15) * DMODEL;
  const __bf16* a1 = a0 + (size_t)16 * DMODEL;
  const __bf16* br[4];
#pragma unroll
  for (int t = 0; t < 4; ++t)
    br[t] = Wt + (size_t)(tn * 64 + t * 16 + l15) * DMODEL;

  gemm32x64(a0, a1, br, off0, acc);

  const int rowhalf = (lane >> 4) * 8;
#pragma unroll
  for (int t = 0; t < 4; ++t) {
    const int col = tn * 64 + t * 16 + l15;
    const float bv = bias[col];
#pragma unroll
    for (int g = 0; g < 2; ++g) {
#pragma unroll
      for (int r = 0; r < 8; ++r) {
        const int row = tm * 32 + g * 16 + rowhalf + r;
        const float v = acc[g][t][r] + bv;
        const int b  = row >> 11;       // /2048
        const int nq = row & 2047;
        if (col < DMODEL) {             // Q (fold in 1/sqrt(dh) = 0.125)
          const int f = col;
          Qb[((size_t)(b * NHEAD + (f >> 6)) * SEQ + nq) * DHEAD + (f & 63)] = (__bf16)(v * 0.125f);
        } else if (col < 2 * DMODEL) {  // K
          const int f = col - DMODEL;
          Kb[((size_t)(b * NHEAD + (f >> 6)) * SEQ + nq) * DHEAD + (f & 63)] = (__bf16)v;
        } else {                        // V stored transposed: [bh][dh][n]
          const int f = col - 2 * DMODEL;
          Vt[((size_t)(b * NHEAD + (f >> 6)) * DHEAD + (f & 63)) * SEQ + nq] = (__bf16)v;
        }
      }
    }
  }
}

// ---------------- flash attention: 1 wave = 16 query rows of one (b,h) ----------------
__global__ __launch_bounds__(256) void flash_attn(
    const __bf16* __restrict__ Qb, const __bf16* __restrict__ Kb,
    const __bf16* __restrict__ Vt, __bf16* __restrict__ Ab) {
  __shared__ __align__(16) __bf16 Plds[8][16 * 32];   // P tile per wave (1 KB each)

  const int lane = threadIdx.x & 31;
  const int w    = threadIdx.x >> 5;
  const int gw   = blockIdx.x * 8 + w;
  const int bh   = gw >> 7;                 // 0..63
  const int qblk = gw & 127;                // 16-row query block
  const int b    = bh >> 4;
  const int h    = bh & 15;
  const int q0   = qblk * 16;
  const int l15  = lane & 15;
  const int off0 = (lane < 16) ? 0 : 8;
  const int rowhalf = (lane >> 4) * 8;

  const __bf16* Qh  = Qb + (size_t)bh * SEQ * DHEAD;
  const __bf16* Kh  = Kb + (size_t)bh * SEQ * DHEAD;
  const __bf16* Vth = Vt + (size_t)bh * DHEAD * SEQ;

  // Q fragments for this wave's 16 rows (K = 64 -> two 32-wide fragments)
  const __bf16* qrow = Qh + (size_t)(q0 + l15) * DHEAD;
  const v16bf qa0 = ldfrag(qrow, off0);
  const v16bf qa1 = ldfrag(qrow + 32, off0);

  float mrun[8], lrun[8];
#pragma unroll
  for (int r = 0; r < 8; ++r) { mrun[r] = -__builtin_inff(); lrun[r] = 0.0f; }
  v8f o[4];
#pragma unroll
  for (int t = 0; t < 4; ++t) o[t] = vzero8();

  const int nkb = (q0 >> 5) + 1;            // causal: key blocks of 32
  for (int kb = 0; kb < nkb; ++kb) {
    const int kbase = kb * 32;

    // ---- S = Q K^T  (16x32 tile as two 16x16 f32 accumulators) ----
    v8f s[2];
#pragma unroll
    for (int nh = 0; nh < 2; ++nh) {
      const __bf16* kr = Kh + (size_t)(kbase + nh * 16 + l15) * DHEAD;
      v8f z = vzero8();
      z = wmma_bf16(qa0, ldfrag(kr, off0), z);
      s[nh] = wmma_bf16(qa1, ldfrag(kr + 32, off0), z);
    }

    // ---- causal mask + online softmax (row = r + rowhalf spans 16 lanes of a half) ----
    float pm[2][8];
#pragma unroll
    for (int r = 0; r < 8; ++r) {
      const int qrowg = q0 + rowhalf + r;
#pragma unroll
      for (int nh = 0; nh < 2; ++nh) {
        const int kcol = kbase + nh * 16 + l15;
        if (kcol > qrowg) s[nh][r] = -__builtin_inff();
      }
      float vmx = fmaxf(s[0][r], s[1][r]);
#pragma unroll
      for (int m = 1; m < 16; m <<= 1) vmx = fmaxf(vmx, __shfl_xor(vmx, m, 32));
      const float mnew  = fmaxf(mrun[r], vmx);
      const float alpha = __expf(mrun[r] - mnew);
      mrun[r] = mnew;
      float ls = 0.0f;
#pragma unroll
      for (int nh = 0; nh < 2; ++nh) {
        const float p = __expf(s[nh][r] - mnew);
        pm[nh][r] = p;
        ls += p;
      }
#pragma unroll
      for (int m = 1; m < 16; m <<= 1) ls += __shfl_xor(ls, m, 32);
      lrun[r] = lrun[r] * alpha + ls;
#pragma unroll
      for (int t = 0; t < 4; ++t) o[t][r] *= alpha;
    }

    // ---- C-layout P -> LDS (row major 16x32 bf16) -> A fragment ----
    __bf16* pl = &Plds[w][0];
#pragma unroll
    for (int nh = 0; nh < 2; ++nh)
#pragma unroll
      for (int r = 0; r < 8; ++r)
        pl[(rowhalf + r) * 32 + nh * 16 + l15] = (__bf16)pm[nh][r];
    asm volatile("s_wait_dscnt 0" ::: "memory");   // wave-local LDS transpose hand-off
    const v16bf pa = ldfrag(pl + l15 * 32, off0);

    // ---- O += P @ V  (V^T rows are contiguous in K) ----
#pragma unroll
    for (int t = 0; t < 4; ++t) {
      const __bf16* vr = Vth + (size_t)(t * 16 + l15) * SEQ + kbase;
      o[t] = wmma_bf16(pa, ldfrag(vr, off0), o[t]);
    }
  }

  // ---- normalize and store to [8192,1024] bf16 for the output projection ----
#pragma unroll
  for (int r = 0; r < 8; ++r) {
    const float inv = 1.0f / lrun[r];
    const int rowg = b * SEQ + q0 + rowhalf + r;
#pragma unroll
    for (int t = 0; t < 4; ++t)
      Ab[(size_t)rowg * DMODEL + h * DHEAD + t * 16 + l15] = (__bf16)(o[t][r] * inv);
  }
}

// ---------------- output projection: [8192,1024] @ Wout + b_out -> fp32 ----------------
__global__ __launch_bounds__(256) void out_gemm(
    const __bf16* __restrict__ A, const __bf16* __restrict__ Wt,
    const float* __restrict__ bias, float* __restrict__ out) {
  const int lane = threadIdx.x & 31;
  const int gw   = blockIdx.x * 8 + (threadIdx.x >> 5);
  const int TN   = DMODEL / 64;               // 16
  const int tm   = gw / TN, tn = gw - tm * TN;
  const int l15  = lane & 15;
  const int off0 = (lane < 16) ? 0 : 8;

  v8f acc[2][4];
#pragma unroll
  for (int g = 0; g < 2; ++g)
#pragma unroll
    for (int t = 0; t < 4; ++t) acc[g][t] = vzero8();

  const __bf16* a0 = A + (size_t)(tm * 32 + l15) * DMODEL;
  const __bf16* a1 = a0 + (size_t)16 * DMODEL;
  const __bf16* br[4];
#pragma unroll
  for (int t = 0; t < 4; ++t)
    br[t] = Wt + (size_t)(tn * 64 + t * 16 + l15) * DMODEL;

  gemm32x64(a0, a1, br, off0, acc);

  const int rowhalf = (lane >> 4) * 8;
#pragma unroll
  for (int t = 0; t < 4; ++t) {
    const int col = tn * 64 + t * 16 + l15;
    const float bv = bias[col];
#pragma unroll
    for (int g = 0; g < 2; ++g)
#pragma unroll
      for (int r = 0; r < 8; ++r) {
        const int row = tm * 32 + g * 16 + rowhalf + r;
        out[(size_t)row * DMODEL + col] = acc[g][t][r] + bv;
      }
  }
}

// ---------------- host launcher ----------------
extern "C" void kernel_launch(void* const* d_in, const int* in_sizes, int n_in,
                              void* d_out, int out_size, void* d_ws, size_t ws_size,
                              hipStream_t stream) {
  const float* x     = (const float*)d_in[0];
  const float* Wqkv  = (const float*)d_in[1];
  const float* bqkv  = (const float*)d_in[2];
  const float* Wout  = (const float*)d_in[3];
  const float* bout  = (const float*)d_in[4];
  float* out = (float*)d_out;

  char* ws = (char*)d_ws;
  __bf16* Xbf    = (__bf16*)(ws + OFF_X);
  __bf16* Wqkvt  = (__bf16*)(ws + OFF_WQKVT);
  __bf16* Woutt  = (__bf16*)(ws + OFF_WOUTT);
  __bf16* Qb     = (__bf16*)(ws + OFF_Q);
  __bf16* Kb     = (__bf16*)(ws + OFF_K);
  __bf16* Vt     = (__bf16*)(ws + OFF_V);
  __bf16* Ab     = (__bf16*)(ws + OFF_A);

  // prep: convert X, transpose+convert weights
  {
    int n = MROWS * DMODEL;
    cvt_f32_bf16<<<(n + 255) / 256, 256, 0, stream>>>(x, Xbf, n);
  }
  {
    int n = NQKV * DMODEL;
    transpose_bf16<<<(n + 255) / 256, 256, 0, stream>>>(Wqkv, Wqkvt, DMODEL, NQKV, n);
  }
  {
    int n = DMODEL * DMODEL;
    transpose_bf16<<<(n + 255) / 256, 256, 0, stream>>>(Wout, Woutt, DMODEL, DMODEL, n);
  }

  // QKV projection + per-head scatter: (8192/32)*(3072/64) = 12288 waves / 8 per block
  qkv_gemm<<<1536, 256, 0, stream>>>(Xbf, Wqkvt, bqkv, Qb, Kb, Vt);

  // flash attention: 64 (b,h) * 128 query blocks = 8192 waves / 8 per block
  flash_attn<<<1024, 256, 0, stream>>>(Qb, Kb, Vt, Ab);

  // output projection: (8192/32)*(1024/64) = 4096 waves / 8 per block
  out_gemm<<<512, 256, 0, stream>>>(Ab, Woutt, bout, out);
}
